// GlobalAttentionGeneral_687194768318
// MI455X (gfx1250) — matrix operand order repaired and checked by reading, not compiled, not run on hardware
//
#include <hip/hip_runtime.h>
#include <hip/hip_bf16.h>

// ---------------------------------------------------------------------------
// GlobalAttentionGeneral on MI455X (gfx1250): 3-stage bf16 WMMA pipeline.
//   Stage 1: sourceT = W x context     (bf16 WMMA, f32 acc; writes sourceT in
//            BOTH row-major [i][s] and transposed [s][i] bf16 layouts)
//   Stage 2: attn = target^T x sourceT + masked softmax (f32); B-fragments
//            come straight from the transposed copy as global b128 loads
//   Stage 3: wctx = sourceT x attn^T   (direct b128 fragment loads, no LDS)
// Memory-bound problem (~0.74 GB min traffic -> ~32us @ 23.3 TB/s); all
// f32->bf16 conversion uses native casts fused into the GEMM load paths, and
// every WMMA fragment is assembled from b128 loads (no element gathers).
// ---------------------------------------------------------------------------

typedef __attribute__((ext_vector_type(16))) __bf16 v16bf;
typedef __attribute__((ext_vector_type(8)))  float  v8f;

#define BB    32
#define IDF_  1024
#define CDF_  1024
#define QQ    2304     // 48*48
#define SS    256

__device__ __forceinline__ __bf16 f2b(float f) { return (__bf16)f; }  // native cvt

union Frag16 { uint4 u4[2]; v16bf v; };       // 32B: one 16x32 bf16 fragment per lane
union Pack8  { __bf16 h[8]; uint4 u; };       // 8 bf16 packed for one b128 store

#define WMMA_BF16(A, B, C) \
  __builtin_amdgcn_wmma_f32_16x16x32_bf16(false, (A), false, (B), (short)0, (C), false, false)

// ---------------------------------------------------------------------------
// Kernel 1: sourceT[b][i][s] = sum_k W[i][k] * context[b][k][s]
// Block: 256 thr (8 waves as 4x2), tile 128(M) x 64(N), K-step 32.
// B tile lives in LDS in fragment-native layout [n-tile][lane][e] so every
// fragment (A and B) is exactly two ds_load_b128.
// ---------------------------------------------------------------------------
__global__ __launch_bounds__(256) void k_gemm_source(
    const float* __restrict__ Wm, const float* __restrict__ ctx,
    __bf16* __restrict__ srcT, __bf16* __restrict__ srcTt)
{
  __shared__ __bf16 ldsA[128][40];       // W tile [m][k], 80B rows (b128-aligned)
  __shared__ __bf16 fragB[4][32][16];    // context tile, fragment-native (4 KB)

  const int b    = blockIdx.z;
  const int i0   = blockIdx.x * 128;
  const int s0   = blockIdx.y * 64;
  const int tid  = threadIdx.x;
  const int lane = tid & 31, wid = tid >> 5;
  const int mbase = (wid >> 1) * 32, nbase = (wid & 1) * 32;
  const int halfA = lane >> 4, ml = lane & 15;
  const int nl = lane & 15;

  const float* ctxb = ctx + (size_t)b * CDF_ * SS;

  const v8f vzero = {0.f,0.f,0.f,0.f,0.f,0.f,0.f,0.f};
  v8f acc[2][2] = {{vzero, vzero}, {vzero, vzero}};

  for (int k0 = 0; k0 < CDF_; k0 += 32) {
    __syncthreads();
    // A = W[i0..i0+127][k0..k0+31], convert + stage row-major
    #pragma unroll
    for (int it = 0; it < 4; ++it) {
      int idx = (tid + it * 256) * 4;
      int row = idx >> 5, col = idx & 31;
      float4 v = *(const float4*)(Wm + (size_t)(i0 + row) * CDF_ + k0 + col);
      ldsA[row][col+0] = f2b(v.x); ldsA[row][col+1] = f2b(v.y);
      ldsA[row][col+2] = f2b(v.z); ldsA[row][col+3] = f2b(v.w);
    }
    // B = context[b][k0..k0+31][s0..s0+63], convert + scatter to fragment layout
    // fragment decode: lane = (s&15) | ((k>>4)<<4), element e = k&15
    #pragma unroll
    for (int it = 0; it < 2; ++it) {
      int idx = (tid + it * 256) * 4;
      int row = idx >> 6, col = idx & 63;        // row = k, col = s (mult of 4)
      float4 v = *(const float4*)(ctxb + (size_t)(k0 + row) * SS + s0 + col);
      const int tb = col >> 4, lhi = (row >> 4) << 4, e = row & 15;
      fragB[tb][lhi | ((col + 0) & 15)][e] = f2b(v.x);
      fragB[tb][lhi | ((col + 1) & 15)][e] = f2b(v.y);
      fragB[tb][lhi | ((col + 2) & 15)][e] = f2b(v.z);
      fragB[tb][lhi | ((col + 3) & 15)][e] = f2b(v.w);
    }
    __syncthreads();

    v16bf afr[2], bfr[2];
    #pragma unroll
    for (int t = 0; t < 2; ++t) {                // A: two b128 ds loads
      const __bf16* rp = &ldsA[mbase + t*16 + ml][0];
      Frag16 u;
      u.u4[0] = *(const uint4*)(rp + halfA * 16);
      u.u4[1] = *(const uint4*)(rp + halfA * 16 + 8);
      afr[t] = u.v;
    }
    #pragma unroll
    for (int t = 0; t < 2; ++t) {                // B: two b128 ds loads
      const __bf16* rp = &fragB[((wid & 1) << 1) + t][lane][0];
      Frag16 u;
      u.u4[0] = *(const uint4*)(rp);
      u.u4[1] = *(const uint4*)(rp + 8);
      bfr[t] = u.v;
    }
    #pragma unroll
    for (int tm = 0; tm < 2; ++tm)
      #pragma unroll
      for (int tn = 0; tn < 2; ++tn)
        acc[tm][tn] = WMMA_BF16(afr[tm], bfr[tn], acc[tm][tn]);
  }

  // epilogue: write sourceT row-major [i][s] (stage-3 A) and transposed [s][i]
  // (stage-2 B), both bf16
  __bf16* outb  = srcT  + (size_t)b * IDF_ * SS;
  __bf16* outtb = srcTt + (size_t)b * SS * IDF_;
  #pragma unroll
  for (int tm = 0; tm < 2; ++tm)
    for (int tn = 0; tn < 2; ++tn)
      for (int r = 0; r < 8; ++r) {
        int m = i0 + mbase + tm*16 + r + ((lane >> 4) << 3);
        int n = s0 + nbase + tn*16 + nl;
        __bf16 hv = f2b(acc[tm][tn][r]);
        outb [(size_t)m * SS   + n] = hv;
        outtb[(size_t)n * IDF_ + m] = hv;
      }
}

// ---------------------------------------------------------------------------
// Kernel 2: attn[b][q][s] = sum_i input[b][i][q] * sourceT[b][i][s], then
// masked softmax over s. Writes attn_out f32 [b][s][q] and attn bf16 [b][q][s].
// Block: 256 thr, one batch x 32 q-rows x all 256 s; wave w owns s-tiles {w,w+8}.
// B-fragments: two global b128 loads from srcTt [s][i] (L2-resident).
// Faithful mask quirk: since Q % B == 0, row (b,q) is masked by mask[q % 32].
// ---------------------------------------------------------------------------
__global__ __launch_bounds__(256) void k_attn_softmax(
    const float* __restrict__ inp, const __bf16* __restrict__ srcTt,
    const unsigned char* __restrict__ mask,
    float* __restrict__ attn_out, __bf16* __restrict__ attnB)
{
  __shared__ __bf16 ldsA[32][40];    // target^T tile [q][k]
  __shared__ float  ldsS[32][256];   // attn scores for softmax

  const int b    = blockIdx.y;
  const int q0   = blockIdx.x * 32;
  const int tid  = threadIdx.x;
  const int lane = tid & 31, wid = tid >> 5;
  const int halfA = lane >> 4, ml = lane & 15;
  const int halfK = lane >> 4, nl = lane & 15;

  const float*  inpb  = inp   + (size_t)b * IDF_ * QQ;
  const __bf16* srctb = srcTt + (size_t)b * SS * IDF_;

  const v8f vzero = {0.f,0.f,0.f,0.f,0.f,0.f,0.f,0.f};
  v8f acc[2][2] = {{vzero, vzero}, {vzero, vzero}};  // [q-tile][s-tile j]

  for (int k0 = 0; k0 < IDF_; k0 += 32) {
    __syncthreads();
    {  // A: transpose input[b][k][q] tile into ldsA[q][k] with convert
      int idx = tid * 4;                       // 32x32 f32 = 1024 elems
      int kk = idx >> 5, qb = idx & 31;
      float4 v = *(const float4*)(inpb + (size_t)(k0 + kk) * QQ + q0 + qb);
      ldsA[qb+0][kk] = f2b(v.x); ldsA[qb+1][kk] = f2b(v.y);
      ldsA[qb+2][kk] = f2b(v.z); ldsA[qb+3][kk] = f2b(v.w);
    }
    __syncthreads();

    v16bf afr[2];
    #pragma unroll
    for (int t = 0; t < 2; ++t) {
      const __bf16* rp = &ldsA[t*16 + ml][0];
      Frag16 u;
      u.u4[0] = *(const uint4*)(rp + halfA * 16);
      u.u4[1] = *(const uint4*)(rp + halfA * 16 + 8);
      afr[t] = u.v;
    }
    v16bf bfr[2];
    #pragma unroll
    for (int j = 0; j < 2; ++j) {              // B: direct global b128 loads
      int st = (wid + j*8) * 16;
      const __bf16* rp = srctb + (size_t)(st + nl) * IDF_ + k0 + halfK * 16;
      Frag16 u;
      u.u4[0] = *(const uint4*)(rp);
      u.u4[1] = *(const uint4*)(rp + 8);
      bfr[j] = u.v;
    }
    #pragma unroll
    for (int qt = 0; qt < 2; ++qt)
      #pragma unroll
      for (int j = 0; j < 2; ++j)
        acc[qt][j] = WMMA_BF16(afr[qt], bfr[j], acc[qt][j]);
  }

  // spill scores to LDS for row-wise softmax
  #pragma unroll
  for (int qt = 0; qt < 2; ++qt)
    for (int j = 0; j < 2; ++j)
      for (int r = 0; r < 8; ++r) {
        int row = qt*16 + r + ((lane >> 4) << 3);
        int col = (wid + j*8)*16 + nl;
        ldsS[row][col] = acc[qt][j][r];
      }
  __syncthreads();

  float*  aout = attn_out + (size_t)b * SS * QQ;
  __bf16* abf  = attnB    + (size_t)b * QQ * SS;
  #pragma unroll
  for (int rr = 0; rr < 4; ++rr) {             // wave handles 4 rows
    int qi = wid * 4 + rr;
    int q  = q0 + qi;
    const unsigned char* mrow = mask + (size_t)(q % BB) * SS;
    float v[8];
    float mx = -3.4e38f;
    #pragma unroll
    for (int j = 0; j < 8; ++j) {
      int s = lane * 8 + j;
      float x = ldsS[qi][s];
      if (mrow[s]) x = -3.0e38f;               // masked -> "-inf" (finite sentinel)
      v[j] = x;
      mx = fmaxf(mx, x);
    }
    #pragma unroll
    for (int off = 16; off > 0; off >>= 1)
      mx = fmaxf(mx, __shfl_xor(mx, off, 32));
    float sum = 0.f;
    #pragma unroll
    for (int j = 0; j < 8; ++j) { v[j] = __expf(v[j] - mx); sum += v[j]; }
    #pragma unroll
    for (int off = 16; off > 0; off >>= 1)
      sum += __shfl_xor(sum, off, 32);
    float inv = 1.0f / sum;
    Pack8 p;
    #pragma unroll
    for (int j = 0; j < 8; ++j) {
      float pv = v[j] * inv;
      int s = lane * 8 + j;
      aout[(size_t)s * QQ + q] = pv;           // attn_out is [b][s][q]
      p.h[j] = f2b(pv);
    }
    *(uint4*)(abf + (size_t)q * SS + lane * 8) = p.u;  // bf16 [b][q][s]
  }
}

// ---------------------------------------------------------------------------
// Kernel 3: wctx[b][i][q] = sum_s sourceT[b][i][s] * attn[b][q][s]
// Both operands K(s)-contiguous bf16 -> direct b128 fragment loads, no LDS.
// Block: 256 thr (4x2 waves), tile 128(M) x 64(N), K = 256 in 8 steps.
// ---------------------------------------------------------------------------
__global__ __launch_bounds__(256) void k_wctx(
    const __bf16* __restrict__ srcT, const __bf16* __restrict__ attnB,
    float* __restrict__ out)
{
  const int b    = blockIdx.z;
  const int i0   = blockIdx.x * 128;
  const int q0   = blockIdx.y * 64;
  const int tid  = threadIdx.x;
  const int lane = tid & 31, wid = tid >> 5;
  const int mbase = (wid >> 1) * 32, nbase = (wid & 1) * 32;
  const int halfA = lane >> 4, ml = lane & 15;
  const int halfK = lane >> 4, nl = lane & 15;

  const __bf16* Ab = srcT  + (size_t)b * IDF_ * SS;
  const __bf16* Bb = attnB + (size_t)b * QQ  * SS;

  const v8f vzero = {0.f,0.f,0.f,0.f,0.f,0.f,0.f,0.f};
  v8f acc[2][2] = {{vzero, vzero}, {vzero, vzero}};

  #pragma unroll
  for (int k0 = 0; k0 < SS; k0 += 32) {
    v16bf afr[2], bfr[2];
    #pragma unroll
    for (int t = 0; t < 2; ++t) {
      const __bf16* rp = Ab + (size_t)(i0 + mbase + t*16 + ml) * SS + k0;
      __builtin_prefetch(rp + 32, 0, 1);       // global_prefetch_b8, next K tile
      Frag16 u;
      u.u4[0] = *(const uint4*)(rp + halfA * 16);
      u.u4[1] = *(const uint4*)(rp + halfA * 16 + 8);
      afr[t] = u.v;
    }
    #pragma unroll
    for (int t = 0; t < 2; ++t) {
      const __bf16* rp = Bb + (size_t)(q0 + nbase + t*16 + nl) * SS + k0 + halfK * 16;
      Frag16 u;
      u.u4[0] = *(const uint4*)(rp);
      u.u4[1] = *(const uint4*)(rp + 8);
      bfr[t] = u.v;
    }
    #pragma unroll
    for (int tm = 0; tm < 2; ++tm)
      #pragma unroll
      for (int tn = 0; tn < 2; ++tn)
        acc[tm][tn] = WMMA_BF16(afr[tm], bfr[tn], acc[tm][tn]);
  }

  float* ob = out + (size_t)b * IDF_ * QQ;
  #pragma unroll
  for (int tm = 0; tm < 2; ++tm)
    for (int tn = 0; tn < 2; ++tn)
      for (int r = 0; r < 8; ++r) {
        int m = i0 + mbase + tm*16 + r + ((lane >> 4) << 3);
        int n = q0 + nbase + tn*16 + nl;
        ob[(size_t)m * QQ + n] = acc[tm][tn][r];
      }
}

// ---------------------------------------------------------------------------
extern "C" void kernel_launch(void* const* d_in, const int* in_sizes, int n_in,
                              void* d_out, int out_size, void* d_ws, size_t ws_size,
                              hipStream_t stream) {
  const float*         input   = (const float*)d_in[0];          // [B,IDF,48,48]
  const float*         context = (const float*)d_in[1];          // [B,CDF,S]
  const unsigned char* mask    = (const unsigned char*)d_in[2];  // [B,S] bool (1B)
  const float*         Wm      = (const float*)d_in[3];          // [IDF,CDF]

  float* wctx_out = (float*)d_out;                               // [B,IDF,48,48]
  float* attn_out = wctx_out + (size_t)BB * IDF_ * QQ;           // [B,S,48,48]

  const size_t srcBytes = (size_t)BB * IDF_ * SS * 2;            // 16 MB each
  __bf16* srcT  = (__bf16*)d_ws;                                 // [b][i][s]
  __bf16* srcTt = (__bf16*)((char*)d_ws + srcBytes);             // [b][s][i]
  __bf16* attnB = (__bf16*)((char*)d_ws + 2 * srcBytes);         // [b][q][s] 36 MB

  dim3 blk(256, 1, 1);

  // Stage 1: sourceT = W x context (writes both layouts)
  k_gemm_source<<<dim3(IDF_ / 128, SS / 64, BB), blk, 0, stream>>>(Wm, context,
                                                                   srcT, srcTt);

  // Stage 2: attn scores + masked softmax
  k_attn_softmax<<<dim3(QQ / 32, BB, 1), blk, 0, stream>>>(input, srcTt, mask,
                                                           attn_out, attnB);

  // Stage 3: weightedContext = sourceT x attn^T
  k_wctx<<<dim3(IDF_ / 128, QQ / 64, BB), blk, 0, stream>>>(srcT, attnB, wctx_out);
}
